// CausalSelfAttention_27693949124777
// MI455X (gfx1250) — matrix operand order repaired and checked by reading, not compiled, not run on hardware
//
#include <hip/hip_runtime.h>
#include <hip/hip_bf16.h>

// ---------------- CDNA5 WMMA types ----------------
typedef __attribute__((ext_vector_type(16))) __bf16 bf16x16;
typedef __attribute__((ext_vector_type(8)))  float  f32x8;

#define T_SEQ  2048
#define C_DIM  1024
#define H_NUM  16
#define D_HEAD 64
#define B_BATCH 2
#define M_ROWS (B_BATCH * T_SEQ)   // 4096
#define BSTRIDE 40                 // padded LDS row stride (elements) to spread banks

__device__ __forceinline__ __bf16 f2bf(float f) {
  union { float f; unsigned u; } x; x.f = f;
  unsigned r = x.u + 0x7FFFu + ((x.u >> 16) & 1u);   // round-to-nearest-even
  union { unsigned short s; __bf16 b; } y;
  y.s = (unsigned short)(r >> 16);
  return y.b;
}

__device__ __forceinline__ bf16x16 mk_frag(uint4 lo, uint4 hi) {
  union { uint4 u[2]; bf16x16 v; } x;
  x.u[0] = lo; x.u[1] = hi;
  return x.v;
}

__device__ __forceinline__ f32x8 wmma_bf16(bf16x16 a, bf16x16 b, f32x8 c) {
  // (neg_a, A, neg_b, B, c_mod, C, reuse_a, reuse_b)
  return __builtin_amdgcn_wmma_f32_16x16x32_bf16(false, a, false, b, (short)0, c,
                                                 false, false);
}

// ---------------- conversion kernels ----------------
__global__ void k_f32_to_bf16(const float* __restrict__ in,
                              __bf16* __restrict__ out, int n) {
  int i = blockIdx.x * blockDim.x + threadIdx.x;
  if (i < n) out[i] = f2bf(in[i]);
}

// W[K,N] (row-major) -> Wt[N,K] bf16 (row-major), so B-fragments load contiguously
__global__ void k_transpose_to_bf16(const float* __restrict__ W,
                                    __bf16* __restrict__ Wt, int K, int N) {
  int i = blockIdx.x * blockDim.x + threadIdx.x;
  if (i >= K * N) return;
  int k = i / N;
  int n = i - k * N;
  Wt[(size_t)n * K + k] = f2bf(W[i]);
}

// ---------------- async stage of one 128x32 B tile into LDS ----------------
// tile layout: [128 cols][BSTRIDE elems] (only first 32 used per col)
__device__ __forceinline__ void stage_btile_async(const __bf16* __restrict__ Bt,
                                                  int Kd, int colBlock, int k0,
                                                  __bf16* dstBase, int tid) {
  const unsigned ldsBase = (unsigned)(uintptr_t)dstBase;  // low 32 bits = LDS offset
#pragma unroll
  for (int it = 0; it < 2; ++it) {
    const int ch = tid + it * 256;          // 512 chunks of 16B = 8 KiB tile
    const int c = ch >> 2;                  // column 0..127
    const int sub = ch & 3;                 // 8-element sub-chunk
    const unsigned long long src =
        (unsigned long long)(uintptr_t)(Bt + (size_t)(colBlock + c) * Kd + k0 + sub * 8);
    const unsigned dst = ldsBase + (unsigned)((c * BSTRIDE + sub * 8) * 2);
    asm volatile("global_load_async_to_lds_b128 %0, %1, off"
                 :: "v"(dst), "v"(src) : "memory");
  }
}

__device__ __forceinline__ void wait_async_all() {
  asm volatile("s_wait_asynccnt 0x0" ::: "memory");
}

// ---------------- shared GEMM core: one wave computes 32(M) x 64(N) ----------------
// A fragments come straight from global (rows shared 2x, L0-friendly);
// B fragments come from an LDS tile staged with double-buffered async copies.
struct Acc8 { union { f32x8 v; float f[8]; } t[2][4]; };

__device__ __forceinline__ void gemm_core_32x64(const __bf16* __restrict__ A,
                                                const __bf16* __restrict__ Bt,
                                                __bf16* __restrict__ Btile, // [2][128*BSTRIDE]
                                                int Kd, int rowBase, int colBlock,
                                                int tid, Acc8& acc) {
  const int lane = tid & 31;
  const int l16 = lane & 15, half = lane >> 4;
  const int waveN = (tid >> 5) & 1;

  const __bf16* aBase[2];
#pragma unroll
  for (int mt = 0; mt < 2; ++mt)
    aBase[mt] = A + (size_t)(rowBase + mt * 16 + l16) * Kd + (half ? 16 : 0);

  // per-wave LDS read base for the 4 N-subtiles
  const int colInTile = waveN * 64;

  f32x8 z = {};
#pragma unroll
  for (int mt = 0; mt < 2; ++mt)
#pragma unroll
    for (int nt = 0; nt < 4; ++nt) acc.t[mt][nt].v = z;

  // prologue: stage k0 = 0 into buffer 0
  stage_btile_async(Bt, Kd, colBlock, 0, Btile, tid);
  wait_async_all();
  __syncthreads();

  int buf = 0;
  for (int k0 = 0; k0 < Kd; k0 += 32) {
    const bool more = (k0 + 32) < Kd;
    // kick off next tile's async copy; it overlaps this tile's WMMAs
    if (more) stage_btile_async(Bt, Kd, colBlock, k0 + 32, Btile + (buf ^ 1) * 128 * BSTRIDE, tid);

    if (more) {
      __builtin_prefetch(aBase[0] + k0 + 64, 0, 3);
    }

    bf16x16 aF[2], bF[4];
#pragma unroll
    for (int mt = 0; mt < 2; ++mt) {
      const __bf16* p = aBase[mt] + k0;
      aF[mt] = mk_frag(*(const uint4*)p, *(const uint4*)(p + 8));
    }
#pragma unroll
    for (int nt = 0; nt < 4; ++nt) {
      const __bf16* p = Btile + buf * 128 * BSTRIDE +
                        (colInTile + nt * 16 + l16) * BSTRIDE + (half ? 16 : 0);
      bF[nt] = mk_frag(*(const uint4*)p, *(const uint4*)(p + 8));
    }
#pragma unroll
    for (int mt = 0; mt < 2; ++mt)
#pragma unroll
      for (int nt = 0; nt < 4; ++nt)
        acc.t[mt][nt].v = wmma_bf16(aF[mt], bF[nt], acc.t[mt][nt].v);

    if (more) {
      wait_async_all();   // own async copies for next tile done
      __syncthreads();    // everyone's copies visible; everyone done reading 'buf'
      buf ^= 1;
    }
  }
}

// ---------------- QKV GEMM: qkv = x @ Wqkv + b, routed into Q/K/Vt head layouts ----
__global__ __launch_bounds__(256) void k_gemm_qkv(
    const __bf16* __restrict__ A,   // [4096, 1024]
    const __bf16* __restrict__ Bt,  // [3072, 1024] (W_qkv transposed)
    const float* __restrict__ bias, // [3072]
    __bf16* __restrict__ Qb,        // [B,H,T,D], pre-scaled by 1/sqrt(D)
    __bf16* __restrict__ Kb,        // [B,H,T,D]
    __bf16* __restrict__ Vt)        // [B,H,D,T]
{
  __shared__ __bf16 Btile[2][128 * BSTRIDE];
  const int tid = threadIdx.x;
  const int lane = tid & 31;
  const int wave = tid >> 5;
  const int l16 = lane & 15, half = lane >> 4;
  const int rowBase = blockIdx.y * 128 + (wave >> 1) * 32;
  const int colBlock = blockIdx.x * 128;
  const int colBase = colBlock + (wave & 1) * 64;

  Acc8 acc;
  gemm_core_32x64(A, Bt, &Btile[0][0], C_DIM, rowBase, colBlock, tid, acc);

#pragma unroll
  for (int mt = 0; mt < 2; ++mt)
#pragma unroll
    for (int nt = 0; nt < 4; ++nt)
#pragma unroll
      for (int r = 0; r < 8; ++r) {
        const int row = rowBase + mt * 16 + half * 8 + r;  // token index
        const int col = colBase + nt * 16 + l16;           // 0..3071
        float v = acc.t[mt][nt].f[r] + bias[col];
        const int which = col >> 10;        // 0=Q 1=K 2=V
        const int c = col & 1023;
        const int h = c >> 6, d = c & 63;
        const int b = row >> 11, t = row & (T_SEQ - 1);
        const size_t bh = (size_t)(b * H_NUM + h);
        if (which == 0)
          Qb[(bh * T_SEQ + t) * D_HEAD + d] = f2bf(v * 0.125f);  // 1/sqrt(64)
        else if (which == 1)
          Kb[(bh * T_SEQ + t) * D_HEAD + d] = f2bf(v);
        else
          Vt[(bh * D_HEAD + d) * T_SEQ + t] = f2bf(v);
      }
}

// ---------------- Flash attention: one wave per 16-row query tile ----------------
__global__ __launch_bounds__(128) void k_attn(const __bf16* __restrict__ Qb,
                                              const __bf16* __restrict__ Kb,
                                              const __bf16* __restrict__ Vt,
                                              __bf16* __restrict__ Yb) {
  __shared__ __bf16 pLds[4][16 * 32];  // per-wave private P staging (no barrier needed)
  const int lane = threadIdx.x & 31;
  const int wave = threadIdx.x >> 5;
  const int l16 = lane & 15, half = lane >> 4;

  const int task = blockIdx.x * 4 + wave;     // 0..4095
  const int qt = task & (T_SEQ / 16 - 1);     // 128 q-tiles per head
  const int bh = task >> 7;                   // 0..31
  const int q0 = qt * 16;

  const __bf16* Qh = Qb + (size_t)bh * T_SEQ * D_HEAD;
  const __bf16* Kh = Kb + (size_t)bh * T_SEQ * D_HEAD;
  const __bf16* Vh = Vt + (size_t)bh * D_HEAD * T_SEQ;
  __bf16* myP = pLds[wave];

  // Q fragments (A layout), constant over the key loop
  const __bf16* qp = Qh + (size_t)(q0 + l16) * D_HEAD + (half ? 16 : 0);
  const bf16x16 qF0 = mk_frag(*(const uint4*)qp,        *(const uint4*)(qp + 8));
  const bf16x16 qF1 = mk_frag(*(const uint4*)(qp + 32), *(const uint4*)(qp + 40));

  union Acc { f32x8 v; float f[8]; };
  Acc o[4];
  f32x8 z = {};
#pragma unroll
  for (int nd = 0; nd < 4; ++nd) o[nd].v = z;
  float mrow[8], lrow[8];
#pragma unroll
  for (int r = 0; r < 8; ++r) { mrow[r] = -3.0e38f; lrow[r] = 0.f; }

  const int rowg0 = q0 + half * 8;  // C-layout: row = VGPR idx + 8*(lane>=16)

  for (int j0 = 0; j0 <= q0 + 15; j0 += 32) {
    // ---- S = Q @ K^T for 32 keys (2 x 16-key subtiles) ----
    Acc s[2];
#pragma unroll
    for (int nj = 0; nj < 2; ++nj) {
      const __bf16* kp = Kh + (size_t)(j0 + nj * 16 + l16) * D_HEAD + (half ? 16 : 0);
      const bf16x16 k0f = mk_frag(*(const uint4*)kp,        *(const uint4*)(kp + 8));
      const bf16x16 k1f = mk_frag(*(const uint4*)(kp + 32), *(const uint4*)(kp + 40));
      f32x8 a = z;
      a = wmma_bf16(qF0, k0f, a);
      a = wmma_bf16(qF1, k1f, a);
      s[nj].v = a;
    }
    // ---- causal mask + online softmax (per-row, half-wave reductions) ----
#pragma unroll
    for (int r = 0; r < 8; ++r) {
      const int rowg = rowg0 + r;
#pragma unroll
      for (int nj = 0; nj < 2; ++nj) {
        const int keyg = j0 + nj * 16 + l16;
        if (keyg > rowg) s[nj].f[r] = -3.0e38f;
      }
      float mx = fmaxf(s[0].f[r], s[1].f[r]);
#pragma unroll
      for (int sh = 8; sh >= 1; sh >>= 1) mx = fmaxf(mx, __shfl_xor(mx, sh, 32));
      const float mnew = fmaxf(mrow[r], mx);
      const float corr = __expf(mrow[r] - mnew);
      const float p0 = __expf(s[0].f[r] - mnew);
      const float p1 = __expf(s[1].f[r] - mnew);
      s[0].f[r] = p0; s[1].f[r] = p1;
      float ssum = p0 + p1;
#pragma unroll
      for (int sh = 8; sh >= 1; sh >>= 1) ssum += __shfl_xor(ssum, sh, 32);
      lrow[r] = lrow[r] * corr + ssum;
      mrow[r] = mnew;
#pragma unroll
      for (int nd = 0; nd < 4; ++nd) o[nd].f[r] *= corr;
    }
    // ---- P -> LDS (row-major 16x32), re-read in A-fragment layout ----
    // single-wave DS ops are architecturally in-order: no barrier required
#pragma unroll
    for (int nj = 0; nj < 2; ++nj)
#pragma unroll
      for (int r = 0; r < 8; ++r)
        myP[(half * 8 + r) * 32 + nj * 16 + l16] = f2bf(s[nj].f[r]);
    const __bf16* pp = myP + l16 * 32 + (half ? 16 : 0);
    const bf16x16 pF = mk_frag(*(const uint4*)pp, *(const uint4*)(pp + 8));
    // ---- O += P @ V  (B-fragment contiguous thanks to V transpose) ----
#pragma unroll
    for (int nd = 0; nd < 4; ++nd) {
      const __bf16* vp = Vh + (size_t)(nd * 16 + l16) * T_SEQ + j0 + (half ? 16 : 0);
      const bf16x16 vF = mk_frag(*(const uint4*)vp, *(const uint4*)(vp + 8));
      o[nd].v = wmma_bf16(pF, vF, o[nd].v);
    }
  }
  // ---- epilogue: Y[b,t,h*64+d] = O/l ----
  const int b = bh >> 4, h = bh & 15;
#pragma unroll
  for (int nd = 0; nd < 4; ++nd)
#pragma unroll
    for (int r = 0; r < 8; ++r) {
      const int rowg = rowg0 + r;
      const int d = nd * 16 + l16;
      Yb[((size_t)(b * T_SEQ + rowg)) * C_DIM + h * D_HEAD + d] =
          f2bf(o[nd].f[r] / lrow[r]);
    }
}

// ---------------- projection GEMM: out = Y @ Wproj + b (f32 out) ----------------
__global__ __launch_bounds__(256) void k_gemm_proj(
    const __bf16* __restrict__ A,   // [4096, 1024]
    const __bf16* __restrict__ Bt,  // [1024, 1024] (W_proj transposed)
    const float* __restrict__ bias, // [1024]
    float* __restrict__ out)        // [4096, 1024]
{
  __shared__ __bf16 Btile[2][128 * BSTRIDE];
  const int tid = threadIdx.x;
  const int lane = tid & 31;
  const int wave = tid >> 5;
  const int l16 = lane & 15, half = lane >> 4;
  const int rowBase = blockIdx.y * 128 + (wave >> 1) * 32;
  const int colBlock = blockIdx.x * 128;
  const int colBase = colBlock + (wave & 1) * 64;

  Acc8 acc;
  gemm_core_32x64(A, Bt, &Btile[0][0], C_DIM, rowBase, colBlock, tid, acc);

#pragma unroll
  for (int mt = 0; mt < 2; ++mt)
#pragma unroll
    for (int nt = 0; nt < 4; ++nt)
#pragma unroll
      for (int r = 0; r < 8; ++r) {
        const int row = rowBase + mt * 16 + half * 8 + r;
        const int col = colBase + nt * 16 + l16;
        out[(size_t)row * C_DIM + col] = acc.t[mt][nt].f[r] + bias[col];
      }
}

// ---------------- host entry ----------------
extern "C" void kernel_launch(void* const* d_in, const int* in_sizes, int n_in,
                              void* d_out, int out_size, void* d_ws, size_t ws_size,
                              hipStream_t stream) {
  (void)in_sizes; (void)n_in; (void)out_size; (void)ws_size;
  const float* x     = (const float*)d_in[0];
  const float* Wqkv  = (const float*)d_in[1];
  const float* bqkv  = (const float*)d_in[2];
  const float* Wproj = (const float*)d_in[3];
  const float* bproj = (const float*)d_in[4];
  float* out = (float*)d_out;

  char* ws = (char*)d_ws;
  __bf16* xb     = (__bf16*)(ws + 0);          //  8 MiB  [4096,1024]
  __bf16* Wqkvt  = (__bf16*)(ws + 8388608);    //  6 MiB  [3072,1024]
  __bf16* Wprojt = (__bf16*)(ws + 14680064);   //  2 MiB  [1024,1024]
  __bf16* Qb     = (__bf16*)(ws + 16777216);   //  8 MiB  [B,H,T,D]
  __bf16* Kb     = (__bf16*)(ws + 25165824);   //  8 MiB  [B,H,T,D]
  __bf16* Vt     = (__bf16*)(ws + 33554432);   //  8 MiB  [B,H,D,T]
  __bf16* Yb     = (__bf16*)(ws + 41943040);   //  8 MiB  [4096,1024]

  k_f32_to_bf16<<<(M_ROWS * C_DIM) / 256, 256, 0, stream>>>(x, xb, M_ROWS * C_DIM);
  k_transpose_to_bf16<<<(C_DIM * 3 * C_DIM) / 256, 256, 0, stream>>>(Wqkv, Wqkvt,
                                                                     C_DIM, 3 * C_DIM);
  k_transpose_to_bf16<<<(C_DIM * C_DIM) / 256, 256, 0, stream>>>(Wproj, Wprojt,
                                                                 C_DIM, C_DIM);
  k_gemm_qkv<<<dim3(24, 32), 256, 0, stream>>>(xb, Wqkvt, bqkv, Qb, Kb, Vt);
  k_attn<<<dim3(1024), 128, 0, stream>>>(Qb, Kb, Vt, Yb);
  k_gemm_proj<<<dim3(8, 32), 256, 0, stream>>>(Yb, Wprojt, bproj, out);
}
